// GCN_fusion2_91036126806361
// MI455X (gfx1250) — compile-verified, matrix operand order checked by reading, not run on hardware
//
#include <hip/hip_runtime.h>
#include <hip/hip_bf16.h>

// ---------------------------------------------------------------------------
// GCN fusion for MI455X (gfx1250, wave32).
//  - Dense GEMMs: bf16 V_WMMA_F32_16X16X32_BF16, fp32 accumulate.
//  - SpMM: edge-parallel, float4 gathers (L2-resident), f32 global atomics.
//  - Pool: LDS-reduced column sums.  Head: single 64-thread block.
// ---------------------------------------------------------------------------

#define N_NODES 50000
#define N_EDGES 800000
#define NFEAT   128
#define NHID    96
#define NCLASS  64
#define NEXT    32

typedef __attribute__((ext_vector_type(16))) __bf16 v16bf;
typedef __attribute__((ext_vector_type(8)))  float  v8f;

__device__ __forceinline__ float selu_f(float x) {
    const float scale = 1.0507009873554805f;
    const float alpha = 1.6732632423543772f;
    return x > 0.f ? scale * x : scale * alpha * (__expf(x) - 1.f);
}

// ---------------------------------------------------------------------------
// C[M,NOUT] = A[M,K] @ B[K,NOUT], row-major fp32 in/out, bf16 WMMA inside.
// One wave computes a 16x16 tile; 4 waves/block stride over M-tiles.
// M must be a multiple of 16 (50000 = 3125*16).
// ---------------------------------------------------------------------------
template <int K, int NOUT>
__global__ void wmma_gemm_bf16(const float* __restrict__ A,
                               const float* __restrict__ B,
                               float* __restrict__ C, int M) {
    const int lane   = threadIdx.x & 31;
    const int wid    = threadIdx.x >> 5;
    const int mtiles = M >> 4;
    const int mt     = blockIdx.x * 4 + wid;   // wave-uniform
    if (mt >= mtiles) return;                  // whole-wave exit, EXEC stays full
    const int nt     = blockIdx.y;
    const int lane16 = lane & 15;
    const int hs     = lane >> 4;              // 0: K 0..7/16..23, 1: K 8..15/24..31

    const float* arow = A + (size_t)(mt * 16 + lane16) * K;
    const float* bcol = B + nt * 16 + lane16;  // stride NOUT down K

    v8f acc = {};
#pragma unroll
    for (int kk = 0; kk < K; kk += 32) {
        const int kb = kk + hs * 8;
        // A fragment: 16 consecutive-ish fp32 -> bf16 (two 8-float runs)
        float4 a0 = *(const float4*)(arow + kb);
        float4 a1 = *(const float4*)(arow + kb + 4);
        float4 a2 = *(const float4*)(arow + kb + 16);
        float4 a3 = *(const float4*)(arow + kb + 20);
        v16bf af;
        af[0]  = (__bf16)a0.x; af[1]  = (__bf16)a0.y; af[2]  = (__bf16)a0.z; af[3]  = (__bf16)a0.w;
        af[4]  = (__bf16)a1.x; af[5]  = (__bf16)a1.y; af[6]  = (__bf16)a1.z; af[7]  = (__bf16)a1.w;
        af[8]  = (__bf16)a2.x; af[9]  = (__bf16)a2.y; af[10] = (__bf16)a2.z; af[11] = (__bf16)a2.w;
        af[12] = (__bf16)a3.x; af[13] = (__bf16)a3.y; af[14] = (__bf16)a3.z; af[15] = (__bf16)a3.w;
        // B fragment: column lane16 of B, same K striping (B is small & cached)
        v16bf bfr;
#pragma unroll
        for (int j = 0; j < 8; ++j) bfr[j]     = (__bf16)bcol[(size_t)(kb + j) * NOUT];
#pragma unroll
        for (int j = 0; j < 8; ++j) bfr[8 + j] = (__bf16)bcol[(size_t)(kb + 16 + j) * NOUT];

        acc = __builtin_amdgcn_wmma_f32_16x16x32_bf16(
            /*neg_a=*/false, af, /*neg_b=*/false, bfr,
            /*c_mod=*/(short)0, acc, /*reuse_a=*/false, /*reuse_b=*/false);
    }

    // C/D layout: VGPR i -> row (i + 8*hs), col lane16
    float* crow = C + (size_t)(mt * 16 + hs * 8) * NOUT + nt * 16 + lane16;
#pragma unroll
    for (int i = 0; i < 8; ++i) crow[(size_t)i * NOUT] = acc[i];
}

// ---------------------------------------------------------------------------
// agg[row[e], :] += edge_w[e] * S[col[e], :]   (F features, float4 vectorized)
// ---------------------------------------------------------------------------
template <int F>
__global__ void spmm_atomic(const int* __restrict__ row, const int* __restrict__ col,
                            const float* __restrict__ ew, const float* __restrict__ S,
                            float* __restrict__ agg, int nEdges) {
    const int G   = F / 4;
    long long idx = (long long)blockIdx.x * blockDim.x + threadIdx.x;
    int e = (int)(idx / G);
    int g = (int)(idx % G);
    if (e >= nEdges) return;
    const float w = ew[e];
    const int   c = col[e];
    const int   r = row[e];
    const float4 v = *(const float4*)(S + (size_t)c * F + g * 4);
    float* dst = agg + (size_t)r * F + g * 4;
    atomicAdd(dst + 0, w * v.x);
    atomicAdd(dst + 1, w * v.y);
    atomicAdd(dst + 2, w * v.z);
    atomicAdd(dst + 3, w * v.w);
}

template <int F>
__global__ void bias_selu_kernel(float* __restrict__ h, const float* __restrict__ bias, int n) {
    int i = blockIdx.x * blockDim.x + threadIdx.x;
    if (i >= n) return;
    h[i] = selu_f(h[i] + bias[i % F]);
}

// Column sums of h[M,64] -> sum[64] (LDS reduce per block, then 1 global atomic/col)
__global__ void colsum64(const float* __restrict__ h, float* __restrict__ sum, int M) {
    __shared__ float ls[64];
    const int t = threadIdx.x;
    if (t < 64) ls[t] = 0.f;
    __syncthreads();
    const int f        = t & 63;
    const int sid      = blockIdx.x * (blockDim.x >> 6) + (t >> 6);
    const int nstreams = gridDim.x * (blockDim.x >> 6);
    float s = 0.f;
    for (int r = sid; r < M; r += nstreams) s += h[(size_t)r * 64 + f];
    atomicAdd(&ls[f], s);
    __syncthreads();
    if (t < 64) atomicAdd(&sum[f], ls[f]);
}

// pooled=selu(mean), bn(sub), z=concat, logits=z@Wf^T+bf, log_softmax, L1(Wf)
__global__ void head_kernel(const float* __restrict__ colsum, const float* __restrict__ sub_fea,
                            const float* __restrict__ Wf, const float* __restrict__ bfv,
                            const float* __restrict__ gamma, const float* __restrict__ beta,
                            const float* __restrict__ mean, const float* __restrict__ var,
                            float* __restrict__ out) {
    __shared__ float z[NCLASS + NEXT];
    __shared__ float logits[NCLASS];
    __shared__ float red[NCLASS];
    __shared__ float lse;
    const int t = threadIdx.x;  // 64 threads
    z[t] = selu_f(colsum[t] * (1.f / (float)N_NODES));
    if (t < NEXT) {
        z[NCLASS + t] = (sub_fea[t] - mean[t]) * rsqrtf(var[t] + 1e-5f) * gamma[t] + beta[t];
    }
    __syncthreads();
    float acc = bfv[t];
    for (int k = 0; k < NCLASS + NEXT; ++k) acc += z[k] * Wf[t * (NCLASS + NEXT) + k];
    logits[t] = acc;
    float l1 = 0.f;
    for (int k = t; k < NCLASS * (NCLASS + NEXT); k += 64) l1 += fabsf(Wf[k]);
    red[t] = l1;
    __syncthreads();
    if (t == 0) {
        float m = logits[0];
        for (int j = 1; j < NCLASS; ++j) m = fmaxf(m, logits[j]);
        float se = 0.f;
        for (int j = 0; j < NCLASS; ++j) se += __expf(logits[j] - m);
        lse = m + logf(se);
        float s = 0.f;
        for (int j = 0; j < 64; ++j) s += red[j];
        out[NCLASS] = s * (1.f / (float)(NCLASS * (NCLASS + NEXT)));
    }
    __syncthreads();
    out[t] = logits[t] - lse;
}

extern "C" void kernel_launch(void* const* d_in, const int* in_sizes, int n_in,
                              void* d_out, int out_size, void* d_ws, size_t ws_size,
                              hipStream_t stream) {
    const float* x   = (const float*)d_in[0];
    const int*   row = (const int*)  d_in[1];
    const int*   col = (const int*)  d_in[2];
    const float* ew  = (const float*)d_in[3];
    const float* sub = (const float*)d_in[4];
    const float* W1  = (const float*)d_in[5];
    const float* b1  = (const float*)d_in[6];
    const float* W2  = (const float*)d_in[7];
    const float* b2  = (const float*)d_in[8];
    const float* Wf  = (const float*)d_in[9];
    const float* bfv = (const float*)d_in[10];
    const float* gam = (const float*)d_in[11];
    const float* bet = (const float*)d_in[12];
    const float* mea = (const float*)d_in[13];
    const float* var = (const float*)d_in[14];
    float* out = (float*)d_out;

    float* bufA   = (float*)d_ws;                        // 50000*96 floats
    float* bufB   = bufA + (size_t)N_NODES * NHID;       // 50000*96 floats
    float* colsum = bufB + (size_t)N_NODES * NHID;       // 64 floats

    // support1 = x @ W1
    {
        dim3 g((N_NODES / 16 + 3) / 4, NHID / 16);
        wmma_gemm_bf16<NFEAT, NHID><<<g, 128, 0, stream>>>(x, W1, bufA, N_NODES);
    }
    // agg1 = spmm(adj, support1); h1 = selu(agg1 + b1)
    hipMemsetAsync(bufB, 0, (size_t)N_NODES * NHID * sizeof(float), stream);
    {
        long long th = (long long)N_EDGES * (NHID / 4);
        spmm_atomic<NHID><<<(int)((th + 255) / 256), 256, 0, stream>>>(row, col, ew, bufA, bufB, N_EDGES);
    }
    bias_selu_kernel<NHID><<<(N_NODES * NHID + 255) / 256, 256, 0, stream>>>(bufB, b1, N_NODES * NHID);

    // support2 = h1 @ W2  (reuse bufA)
    {
        dim3 g((N_NODES / 16 + 3) / 4, NCLASS / 16);
        wmma_gemm_bf16<NHID, NCLASS><<<g, 128, 0, stream>>>(bufB, W2, bufA, N_NODES);
    }
    // agg2 = spmm(adj, support2); h2 = selu(agg2 + b2)  (reuse bufB)
    hipMemsetAsync(bufB, 0, (size_t)N_NODES * NCLASS * sizeof(float), stream);
    {
        long long th = (long long)N_EDGES * (NCLASS / 4);
        spmm_atomic<NCLASS><<<(int)((th + 255) / 256), 256, 0, stream>>>(row, col, ew, bufA, bufB, N_EDGES);
    }
    bias_selu_kernel<NCLASS><<<(N_NODES * NCLASS + 255) / 256, 256, 0, stream>>>(bufB, b2, N_NODES * NCLASS);

    // pooled mean + head
    hipMemsetAsync(colsum, 0, 64 * sizeof(float), stream);
    colsum64<<<256, 256, 0, stream>>>(bufB, colsum, N_NODES);
    head_kernel<<<1, 64, 0, stream>>>(colsum, sub, Wf, bfv, gam, bet, mea, var, out);
}
